// ScaledDotProductAttention_4475355922835
// MI455X (gfx1250) — matrix operand order repaired and checked by reading, not compiled, not run on hardware
//
#include <hip/hip_runtime.h>

#define BATCH 4
#define HEADS 16
#define SEQ   1024
#define DIM   128

constexpr float SCALE  = 0.08838834764831845f; // 1/sqrt(128)
constexpr float NEGBIG = -1e9f;

typedef __attribute__((ext_vector_type(16))) __bf16 v16bf;
typedef __attribute__((ext_vector_type(8)))  float  v8f;
typedef __attribute__((ext_vector_type(4)))  float  v4f;   // native clang vector (NT-load OK)
typedef __attribute__((ext_vector_type(8)))  unsigned int v8u;

// LDS row strides (elements), padded to avoid bank aliasing
#define QS 136   // sQ  rows (ushort)
#define KS 136   // sKV rows (ushort)
#define SS 1028  // sS  rows (float)
#define PS 1032  // sP  rows (ushort)

__device__ __forceinline__ unsigned int f2bf_u(float f) {
    unsigned int u = __builtin_bit_cast(unsigned int, f);
    return (u + 0x7FFFu + ((u >> 16) & 1u)) >> 16;   // RNE
}
__device__ __forceinline__ unsigned int pack2(float lo, float hi) {
    return f2bf_u(lo) | (f2bf_u(hi) << 16);
}
__device__ __forceinline__ unsigned short f2bf(float f) {
    return (unsigned short)f2bf_u(f);
}

// Build a 16-element bf16 fragment from two 16-byte contiguous LDS chunks.
__device__ __forceinline__ v16bf load_frag(const unsigned short* p0,
                                           const unsigned short* p1) {
    const uint4 x0 = *(const uint4*)p0;
    const uint4 x1 = *(const uint4*)p1;
    v8u u;
    u[0] = x0.x; u[1] = x0.y; u[2] = x0.z; u[3] = x0.w;
    u[4] = x1.x; u[5] = x1.y; u[6] = x1.z; u[7] = x1.w;
    return __builtin_bit_cast(v16bf, u);
}

__global__ __launch_bounds__(256, 1)
void attn_fused_kernel(const float* __restrict__ Q, const float* __restrict__ K,
                       const float* __restrict__ V, const int* __restrict__ Msk,
                       float* __restrict__ Out, float* __restrict__ Attn)
{
    __shared__ __align__(16) unsigned short sQ[16 * QS];     //  4352 B : Q tile bf16
    __shared__ __align__(16) unsigned short sKV[128 * KS];   // 34816 B : K chunk / V^T chunk bf16
    __shared__ __align__(16) float          sS[16 * SS];     // 65792 B : scores fp32
    __shared__ __align__(16) unsigned short sP[16 * PS];     // 33024 B : probs bf16

    const int tid   = threadIdx.x;
    const int lane  = tid & 31;
    const int wave  = tid >> 5;                 // 0..7
    const int qt    = blockIdx.x;               // 0..63  (q tile)
    const int bh    = blockIdx.y;               // 0..63  (b*H + h)
    const int b     = bh >> 4;                  // H = 16
    const int qbase = qt * 16;

    const float* Qg = Q + (size_t)bh * SEQ * DIM + (size_t)qbase * DIM;
    const float* Kg = K + (size_t)bh * SEQ * DIM;
    const float* Vg = V + (size_t)bh * SEQ * DIM;
    const int*   Mg = Msk + (size_t)b * SEQ * SEQ + (size_t)qbase * SEQ;
    float*       Og = Out + (size_t)bh * SEQ * DIM + (size_t)qbase * DIM;
    float*       Ag = Attn + (size_t)bh * SEQ * SEQ + (size_t)qbase * SEQ;

    // ---- stage Q tile (16x128 f32 -> bf16 LDS); Q is read-once -> NT load ----
    for (int i = tid; i < 16 * 32; i += 256) {        // 512 float4
        const int r = i >> 5, c4 = i & 31;
        const v4f f = __builtin_nontemporal_load((const v4f*)(Qg + r * DIM + c4 * 4));
        uint2 u; u.x = pack2(f.x, f.y); u.y = pack2(f.z, f.w);
        *(uint2*)&sQ[r * QS + c4 * 4] = u;
    }

    // Per-wave WMMA lane roles (wave32):
    const int m     = lane & 15;                 // A row / C row base
    const int n     = lane & 15;                 // B col / C col
    const int abase = (lane < 16) ? 0 : 8;       // A: lanes16-31 hold K+8..15 / K+24..31
    const int bbase = (lane < 16) ? 0 : 16;      // unused; B fragments load with abase, same k order as A
    const int mrow  = (lane < 16) ? 0 : 8;       // C/D: lanes16-31 hold M = r+8

    // ================= Phase 1: S = (Q K^T) tiles via WMMA =================
    for (int ch = 0; ch < 8; ++ch) {
        __syncthreads();                          // sKV reuse fence
        // stage K rows [ch*128, ch*128+128) as bf16, row-major
        for (int i = tid; i < 128 * 32; i += 256) {   // 4096 float4
            const int r = i >> 5, c4 = i & 31;
            const float4 f = *(const float4*)(Kg + (size_t)(ch * 128 + r) * DIM + c4 * 4);
            uint2 u; u.x = pack2(f.x, f.y); u.y = pack2(f.z, f.w);
            *(uint2*)&sKV[r * KS + c4 * 4] = u;
        }
        // prefetch next K chunk (64 KB) so it lands in cache during WMMA work
        if (ch < 7) {
            const float* nxt = Kg + (size_t)(ch + 1) * 128 * DIM;
            __builtin_prefetch(nxt + (size_t)tid * 64, 0, 1);
            __builtin_prefetch(nxt + (size_t)tid * 64 + 32, 0, 1);
        }
        __syncthreads();

        v8f acc{};
        const unsigned short* brow = &sKV[(wave * 16 + n) * KS];
        #pragma unroll
        for (int kk = 0; kk < 128; kk += 32) {
            const v16bf a  = load_frag(&sQ[m * QS + kk + abase],
                                       &sQ[m * QS + kk + 16 + abase]);
            const v16bf bb = load_frag(brow + kk + abase, brow + kk + 16 + abase);
            acc = __builtin_amdgcn_wmma_f32_16x16x32_bf16(
                false, a, false, bb, (short)0, acc, false, false);
        }
        const int colbase = ch * 128 + wave * 16 + n;
        #pragma unroll
        for (int r = 0; r < 8; ++r)
            sS[(mrow + r) * SS + colbase] = acc[r];
    }
    __syncthreads();

    // ================= Phase 2: masked softmax (fp32, in LDS) =================
    for (int rr = 0; rr < 2; ++rr) {
        const int r = wave * 2 + rr;
        float* srow = &sS[r * SS];
        const int* mr = Mg + (size_t)r * SEQ;

        float mx = NEGBIG;
        for (int c = lane; c < SEQ; c += 32) {
            float v = srow[c] * SCALE;
            if (mr[c] == 0) v = NEGBIG;
            srow[c] = v;
            mx = fmaxf(mx, v);
        }
        #pragma unroll
        for (int off = 16; off > 0; off >>= 1)
            mx = fmaxf(mx, __shfl_xor(mx, off, 32));

        float sum = 0.f;
        for (int c = lane; c < SEQ; c += 32) {
            const float e = __expf(srow[c] - mx);
            srow[c] = e;
            sum += e;
        }
        #pragma unroll
        for (int off = 16; off > 0; off >>= 1)
            sum += __shfl_xor(sum, off, 32);
        const float inv = 1.0f / sum;

        float* arow = Ag + (size_t)r * SEQ;
        unsigned short* prow = &sP[r * PS];
        for (int c = lane; c < SEQ; c += 32) {
            const float p = srow[c] * inv;
            // attention is a 268 MB write-once stream: NT store, keep L2 for K/V/mask
            __builtin_nontemporal_store(p, arow + c);
            prow[c] = f2bf(p);           // bf16 copy for P x V
        }
    }

    // ================= Phase 3: Out = P V via WMMA =================
    v8f oacc{};
    for (int ch = 0; ch < 8; ++ch) {
        __syncthreads();                  // sKV reuse fence (covers softmax on first iter)
        // stage V rows [ch*128, +128) TRANSPOSED: sKV[d][kv_local], bf16
        for (int i = tid; i < 128 * 32; i += 256) {
            const int r = i >> 5, c4 = i & 31;          // r = kv row, c4*4 = d base
            const float4 f = *(const float4*)(Vg + (size_t)(ch * 128 + r) * DIM + c4 * 4);
            const int d0 = c4 * 4;
            sKV[(d0 + 0) * KS + r] = f2bf(f.x);
            sKV[(d0 + 1) * KS + r] = f2bf(f.y);
            sKV[(d0 + 2) * KS + r] = f2bf(f.z);
            sKV[(d0 + 3) * KS + r] = f2bf(f.w);
        }
        // prefetch next V chunk
        if (ch < 7) {
            const float* nxt = Vg + (size_t)(ch + 1) * 128 * DIM;
            __builtin_prefetch(nxt + (size_t)tid * 64, 0, 1);
            __builtin_prefetch(nxt + (size_t)tid * 64 + 32, 0, 1);
        }
        __syncthreads();

        const unsigned short* brow = &sKV[(wave * 16 + n) * KS]; // row = d column
        #pragma unroll
        for (int kk = 0; kk < 128; kk += 32) {
            const int kg = ch * 128 + kk;
            const v16bf a  = load_frag(&sP[m * PS + kg + abase],
                                       &sP[m * PS + kg + 16 + abase]);
            const v16bf bb = load_frag(brow + kk + abase, brow + kk + 16 + abase);
            oacc = __builtin_amdgcn_wmma_f32_16x16x32_bf16(
                false, a, false, bb, (short)0, oacc, false, false);
        }
    }

    #pragma unroll
    for (int r = 0; r < 8; ++r)
        __builtin_nontemporal_store(oacc[r], Og + (size_t)(mrow + r) * DIM + wave * 16 + n);
}

extern "C" void kernel_launch(void* const* d_in, const int* in_sizes, int n_in,
                              void* d_out, int out_size, void* d_ws, size_t ws_size,
                              hipStream_t stream) {
    (void)in_sizes; (void)n_in; (void)d_ws; (void)ws_size; (void)out_size;
    const float* Q   = (const float*)d_in[0];
    const float* K   = (const float*)d_in[1];
    const float* V   = (const float*)d_in[2];
    const int*   Msk = (const int*)d_in[3];

    float* Out  = (float*)d_out;                                   // [B,H,S,D]
    float* Attn = Out + (size_t)BATCH * HEADS * SEQ * DIM;         // [B,H,S,S]

    dim3 grid(SEQ / 16, BATCH * HEADS);
    attn_fused_kernel<<<grid, 256, 0, stream>>>(Q, K, V, Msk, Out, Attn);
}
